// GNN_65532611002728
// MI455X (gfx1250) — matrix operand order repaired and checked by reading, not compiled, not run on hardware
//
#include <hip/hip_runtime.h>
#include <hip/hip_bf16.h>

typedef __bf16 bf16_t;
typedef __attribute__((ext_vector_type(16))) __bf16 v16bf;
typedef __attribute__((ext_vector_type(8)))  __bf16 v8bf;
typedef __attribute__((ext_vector_type(8)))  float  v8f;

#define NEG_SLOPE 0.2f
#define EPS_SM 1e-16f
#define EPS_NORM 1e-12f

// ---------- helpers ----------
__device__ __forceinline__ unsigned short f2bf_raw(float f) {
    unsigned u = __float_as_uint(f);
    return (unsigned short)((u + 0x7fffu + ((u >> 16) & 1u)) >> 16);  // RNE
}
// order-preserving float->u32 encoding (monotone for all finite floats)
__device__ __forceinline__ unsigned f32_ord(float f) {
    unsigned u = __float_as_uint(f);
    return (u >> 31) ? ~u : (u | 0x80000000u);
}
__device__ __forceinline__ float ord_f32(unsigned e) {
    return (e & 0x80000000u) ? __uint_as_float(e & 0x7fffffffu) : __uint_as_float(~e);
}

// ---------- conversion / setup kernels ----------
__global__ void k_f32_to_bf16(const float* __restrict__ in, unsigned short* __restrict__ out, long n) {
    long i = (long)blockIdx.x * blockDim.x + threadIdx.x;
    if (i < n) out[i] = f2bf_raw(in[i]);
}

// W[K][Nout] (row-major) -> WT[Nout][K] bf16
__global__ void k_transpose_bf16(const float* __restrict__ W, unsigned short* __restrict__ WT,
                                 int K, int Nout) {
    int idx = blockIdx.x * blockDim.x + threadIdx.x;
    if (idx >= K * Nout) return;
    int k = idx / Nout, n = idx % Nout;             // coalesced read of W
    WT[(size_t)n * K + k] = f2bf_raw(W[idx]);
}

// ce[h] = sum_c We[h*64+c] * att_e[h*64+c], for both layers
__global__ void k_compute_ce(const float* We1, const float* ae1,
                             const float* We2, const float* ae2, float* ce) {
    int h = threadIdx.x;
    if (h < 4) {
        float s1 = 0.f, s2 = 0.f;
        for (int c = 0; c < 64; ++c) {
            s1 += We1[h * 64 + c] * ae1[h * 64 + c];
            s2 += We2[h * 64 + c] * ae2[h * 64 + c];
        }
        ce[h] = s1; ce[4 + h] = s2;
    }
}

__global__ void k_fill_u32(unsigned* __restrict__ p, unsigned v, long n) {
    long i = (long)blockIdx.x * blockDim.x + threadIdx.x;
    if (i < n) p[i] = v;
}

// ---------- WMMA GEMM: C[M][256] = A[M][K](bf16) * BT[256][K]^T ----------
// block = 256 threads (8 waves); wave w handles N-tiles {2w, 2w+1}; grid.y = M/16
__global__ void k_gemm_bf16_wmma(const bf16_t* __restrict__ A, const bf16_t* __restrict__ BT,
                                 float* __restrict__ C, int M, int K, int Nout) {
    const int lane = threadIdx.x & 31;
    const int wave = threadIdx.x >> 5;
    const int mt   = blockIdx.y;
    const int nt0  = (blockIdx.x * 8 + wave) * 2;
    const int g    = lane >> 4;
    const int ln   = lane & 15;

    const bf16_t* arow  = A  + (size_t)(mt * 16 + ln) * K;
    const bf16_t* brow0 = BT + (size_t)(nt0 * 16 + ln) * K;
    const bf16_t* brow1 = BT + (size_t)((nt0 + 1) * 16 + ln) * K;

    v8f c0 = {}; v8f c1 = {};
    for (int k0 = 0; k0 < K; k0 += 32) {
        union { v16bf v; v8bf h[2]; } af;
        af.v = *(const v16bf*)(arow + k0 + 16 * g);          // 16 contiguous K
        const int kb = k0 + 16 * g;
        v16bf b0 = *(const v16bf*)(brow0 + kb);             // 16 contiguous K
        v16bf b1 = *(const v16bf*)(brow1 + kb);
        c0 = __builtin_amdgcn_wmma_f32_16x16x32_bf16(false, af.v, false, b0, (short)0, c0, false, false);
        c1 = __builtin_amdgcn_wmma_f32_16x16x32_bf16(false, af.v, false, b1, (short)0, c1, false, false);
    }
    const int mrow = mt * 16 + 8 * g;
    for (int v = 0; v < 8; ++v) {
        C[(size_t)(mrow + v) * Nout + nt0 * 16 + ln]       = c0[v];
        C[(size_t)(mrow + v) * Nout + (nt0 + 1) * 16 + ln] = c1[v];
    }
}

// ---------- attention logits: one wave per node (row length 256, H=4, C=64) ----------
__global__ void k_att_logits(const float* __restrict__ xs, const float* __restrict__ att_s,
                             const float* __restrict__ att_d, float* __restrict__ a_s,
                             float* __restrict__ a_d, int Nn) {
    int lane = threadIdx.x & 31;
    int node = blockIdx.x * (blockDim.x >> 5) + (threadIdx.x >> 5);
    if (node >= Nn) return;
    const float* row = xs + (size_t)node * 256;
    int base = lane * 8;
    int h = base >> 6;
    float ps = 0.f, pd = 0.f;
    #pragma unroll
    for (int j = 0; j < 8; ++j) {
        float v = row[base + j];
        ps += v * att_s[base + j];
        pd += v * att_d[base + j];
    }
    #pragma unroll
    for (int off = 1; off < 8; off <<= 1) {   // reduce within each head's 8 lanes
        ps += __shfl_xor(ps, off, 32);
        pd += __shfl_xor(pd, off, 32);
    }
    if ((lane & 7) == 0) {
        a_s[node * 4 + h] = ps;
        a_d[node * 4 + h] = pd;
    }
}

// ---------- edge: alpha = leakyrelu(a_s[src]+a_d[dst]+ew*ce), segment max via atomicMax ----------
__global__ void k_edge_alpha_max(const int* __restrict__ src, const int* __restrict__ dst,
                                 const float* __restrict__ ew, const float* __restrict__ a_s,
                                 const float* __restrict__ a_d, const float* __restrict__ ce,
                                 float* __restrict__ alpha, unsigned* __restrict__ menc, long E) {
    long idx = (long)blockIdx.x * blockDim.x + threadIdx.x;
    if (idx >= E * 4) return;
    long e = idx >> 2; int h = (int)(idx & 3);
    int s = src[e], d = dst[e];
    float a = a_s[(size_t)s * 4 + h] + a_d[(size_t)d * 4 + h] + ew[e] * ce[h];
    a = (a > 0.f) ? a : NEG_SLOPE * a;
    alpha[idx] = a;
    atomicMax(&menc[(size_t)d * 4 + h], f32_ord(a));
}

// ---------- edge: ex = exp(alpha - max[dst]), denom[dst] += ex (in-place on alpha) ----------
__global__ void k_edge_exp_sum(const int* __restrict__ dst, const unsigned* __restrict__ menc,
                               float* __restrict__ alpha_ex, float* __restrict__ den, long E) {
    long idx = (long)blockIdx.x * blockDim.x + threadIdx.x;
    if (idx >= E * 4) return;
    long e = idx >> 2; int h = (int)(idx & 3);
    int d = dst[e];
    float m  = ord_f32(menc[(size_t)d * 4 + h]);
    float ex = expf(alpha_ex[idx] - m);
    alpha_ex[idx] = ex;
    atomicAdd(&den[(size_t)d * 4 + h], ex);
}

// ---------- edge: acc[dst] += xs[src] * ex/(den[dst]+eps); one wave per edge ----------
__global__ void k_edge_message(const int* __restrict__ src, const int* __restrict__ dst,
                               const float* __restrict__ xs, const float* __restrict__ ex,
                               const float* __restrict__ den, float* __restrict__ acc, long E) {
    int lane = threadIdx.x & 31;
    long e = (long)blockIdx.x * (blockDim.x >> 5) + (threadIdx.x >> 5);
    if (e >= E) return;
    int s = src[e], d = dst[e];
    int base = lane * 8;
    int h = base >> 6;
    float w = ex[e * 4 + h] / (den[(size_t)d * 4 + h] + EPS_SM);
    const float4* xr = (const float4*)(xs + (size_t)s * 256 + base);
    float4 v0 = xr[0], v1 = xr[1];
    float* ao = acc + (size_t)d * 256 + base;
    atomicAdd(ao + 0, v0.x * w); atomicAdd(ao + 1, v0.y * w);
    atomicAdd(ao + 2, v0.z * w); atomicAdd(ao + 3, v0.w * w);
    atomicAdd(ao + 4, v1.x * w); atomicAdd(ao + 5, v1.y * w);
    atomicAdd(ao + 6, v1.z * w); atomicAdd(ao + 7, v1.w * w);
}

// ---------- layer1 epilogue: h = relu(l2norm(acc + b1)) -> bf16; one wave per node ----------
__global__ void k_finalize1(const float* __restrict__ acc, const float* __restrict__ b1,
                            unsigned short* __restrict__ hb, int Nn) {
    int lane = threadIdx.x & 31;
    int node = blockIdx.x * (blockDim.x >> 5) + (threadIdx.x >> 5);
    if (node >= Nn) return;
    const float* row = acc + (size_t)node * 256;
    int base = lane * 8;
    float v[8]; float ss = 0.f;
    #pragma unroll
    for (int j = 0; j < 8; ++j) { v[j] = row[base + j] + b1[base + j]; ss += v[j] * v[j]; }
    #pragma unroll
    for (int off = 1; off < 32; off <<= 1) ss += __shfl_xor(ss, off, 32);
    float inv = 1.f / fmaxf(sqrtf(ss), EPS_NORM);
    #pragma unroll
    for (int j = 0; j < 8; ++j) {
        float r = v[j] * inv;
        r = (r > 0.f) ? r : 0.f;
        hb[(size_t)node * 256 + base + j] = f2bf_raw(r);
    }
}

// ---------- layer2 epilogue: out[n][c] = mean_h acc[n][h*64+c] + b2[c] ----------
__global__ void k_finalize2(const float* __restrict__ acc, const float* __restrict__ b2,
                            float* __restrict__ out, int Nn) {
    long idx = (long)blockIdx.x * blockDim.x + threadIdx.x;
    if (idx >= (long)Nn * 64) return;
    int n = (int)(idx >> 6), c = (int)(idx & 63);
    const float* row = acc + (size_t)n * 256;
    out[idx] = 0.25f * (row[c] + row[64 + c] + row[128 + c] + row[192 + c]) + b2[c];
}

// =====================================================================
extern "C" void kernel_launch(void* const* d_in, const int* in_sizes, int n_in,
                              void* d_out, int out_size, void* d_ws, size_t ws_size,
                              hipStream_t stream) {
    const float* x        = (const float*)d_in[0];
    const int*   eidx     = (const int*)d_in[1];
    const float* ew       = (const float*)d_in[2];
    const float* W1       = (const float*)d_in[3];
    const float* att_src1 = (const float*)d_in[4];
    const float* att_dst1 = (const float*)d_in[5];
    const float* We1      = (const float*)d_in[6];
    const float* att_e1   = (const float*)d_in[7];
    const float* b1       = (const float*)d_in[8];
    const float* W2       = (const float*)d_in[9];
    const float* att_src2 = (const float*)d_in[10];
    const float* att_dst2 = (const float*)d_in[11];
    const float* We2      = (const float*)d_in[12];
    const float* att_e2   = (const float*)d_in[13];
    const float* b2       = (const float*)d_in[14];

    const int  Fin = in_sizes[3] / 256;        // 128
    const long Nn  = in_sizes[0] / Fin;        // 50000
    const long E   = in_sizes[1] / 2;          // 800000
    const int* src = eidx;
    const int* dst = eidx + E;

    // ---- workspace layout ----
    char* ws = (char*)d_ws;
    size_t off = 0;
    auto alloc = [&](size_t bytes) { size_t o = off; off = (off + bytes + 255) & ~(size_t)255; return o; };
    size_t O_W1T = alloc((size_t)256 * Fin * 2);
    size_t O_W2T = alloc((size_t)256 * 256 * 2);
    size_t O_CE  = alloc(8 * sizeof(float));
    size_t O_AS  = alloc((size_t)Nn * 4 * 4);
    size_t O_AD  = alloc((size_t)Nn * 4 * 4);
    size_t O_MEN = alloc((size_t)Nn * 4 * 4);
    size_t O_DEN = alloc((size_t)Nn * 4 * 4);
    size_t O_EX  = alloc((size_t)E * 4 * 4);
    size_t O_XB  = alloc((size_t)Nn * Fin * 2);
    size_t O_XS  = alloc((size_t)Nn * 256 * 4);
    size_t O_ACC = alloc((size_t)Nn * 256 * 4);
    size_t O_HB  = alloc((size_t)Nn * 256 * 2);
    (void)ws_size;

    unsigned short* XB  = (unsigned short*)(ws + O_XB);
    unsigned short* W1T = (unsigned short*)(ws + O_W1T);
    unsigned short* W2T = (unsigned short*)(ws + O_W2T);
    unsigned short* HB  = (unsigned short*)(ws + O_HB);
    float*    CE  = (float*)(ws + O_CE);
    float*    AS  = (float*)(ws + O_AS);
    float*    AD  = (float*)(ws + O_AD);
    unsigned* MEN = (unsigned*)(ws + O_MEN);
    float*    DEN = (float*)(ws + O_DEN);
    float*    EX  = (float*)(ws + O_EX);
    float*    XS  = (float*)(ws + O_XS);
    float*    ACC = (float*)(ws + O_ACC);

    const int T = 256;
    auto blocks = [&](long n) { return (unsigned)((n + T - 1) / T); };

    // ---- setup: bf16 conversions, transposed weights, edge constants ----
    k_f32_to_bf16<<<blocks(Nn * Fin), T, 0, stream>>>(x, XB, Nn * Fin);
    k_transpose_bf16<<<blocks((long)Fin * 256), T, 0, stream>>>(W1, W1T, Fin, 256);
    k_transpose_bf16<<<blocks(256L * 256), T, 0, stream>>>(W2, W2T, 256, 256);
    k_compute_ce<<<1, 64, 0, stream>>>(We1, att_e1, We2, att_e2, CE);

    const unsigned wavesPerBlk = T / 32;
    const unsigned nodeBlks = (unsigned)((Nn + wavesPerBlk - 1) / wavesPerBlk);
    const unsigned edgeBlks = (unsigned)((E + wavesPerBlk - 1) / wavesPerBlk);
    dim3 gemmGrid(1, (unsigned)(Nn / 16));

    // ================= layer 1 =================
    k_gemm_bf16_wmma<<<gemmGrid, T, 0, stream>>>((const bf16_t*)XB, (const bf16_t*)W1T, XS,
                                                 (int)Nn, Fin, 256);
    k_att_logits<<<nodeBlks, T, 0, stream>>>(XS, att_src1, att_dst1, AS, AD, (int)Nn);
    k_fill_u32<<<blocks(Nn * 4), T, 0, stream>>>(MEN, 0u, Nn * 4);
    k_fill_u32<<<blocks(Nn * 4), T, 0, stream>>>((unsigned*)DEN, 0u, Nn * 4);
    k_fill_u32<<<blocks(Nn * 256), T, 0, stream>>>((unsigned*)ACC, 0u, Nn * 256);
    k_edge_alpha_max<<<blocks(E * 4), T, 0, stream>>>(src, dst, ew, AS, AD, CE, EX, MEN, E);
    k_edge_exp_sum<<<blocks(E * 4), T, 0, stream>>>(dst, MEN, EX, DEN, E);
    k_edge_message<<<edgeBlks, T, 0, stream>>>(src, dst, XS, EX, DEN, ACC, E);
    k_finalize1<<<nodeBlks, T, 0, stream>>>(ACC, b1, HB, (int)Nn);

    // ================= layer 2 =================
    k_gemm_bf16_wmma<<<gemmGrid, T, 0, stream>>>((const bf16_t*)HB, (const bf16_t*)W2T, XS,
                                                 (int)Nn, 256, 256);
    k_att_logits<<<nodeBlks, T, 0, stream>>>(XS, att_src2, att_dst2, AS, AD, (int)Nn);
    k_fill_u32<<<blocks(Nn * 4), T, 0, stream>>>(MEN, 0u, Nn * 4);
    k_fill_u32<<<blocks(Nn * 4), T, 0, stream>>>((unsigned*)DEN, 0u, Nn * 4);
    k_fill_u32<<<blocks(Nn * 256), T, 0, stream>>>((unsigned*)ACC, 0u, Nn * 256);
    k_edge_alpha_max<<<blocks(E * 4), T, 0, stream>>>(src, dst, ew, AS, AD, CE + 4, EX, MEN, E);
    k_edge_exp_sum<<<blocks(E * 4), T, 0, stream>>>(dst, MEN, EX, DEN, E);
    k_edge_message<<<edgeBlks, T, 0, stream>>>(src, dst, XS, EX, DEN, ACC, E);
    k_finalize2<<<blocks(Nn * 64), T, 0, stream>>>(ACC, b2, (float*)d_out, (int)Nn);
}
